// MultiheadAttention_17300128268327
// MI455X (gfx1250) — compile-verified, matrix operand order checked by reading, not run on hardware
//
#include <hip/hip_runtime.h>
#include <hip/hip_bf16.h>
#include <math.h>

// ---------------------------------------------------------------------------
// MultiheadAttention forward for gfx1250 (wave32, WMMA bf16 path).
// B=2, S=2048, D=1024, H=16, DK=64.
// Pipeline:
//   weight pre-convert fp32 -> bf16 (once)
//   gemm<f32 in, bf16 head-major out>  : Q (scale 1/8 folded), K
//   gemm<f32 in, bf16 transposed out>  : V -> Vt[B][H][64][S]
//   flash attention: async-to-LDS staging, 64-key blocks, double-buffered,
//                    online softmax with WMMA row-sums
//   gemm<bf16 in, f32 row-major out>   : output projection
// ---------------------------------------------------------------------------

typedef __attribute__((ext_vector_type(16))) __bf16 bf16x16;
typedef __attribute__((ext_vector_type(8)))  __bf16 bf16x8;
typedef __attribute__((ext_vector_type(8)))  float  f32x8;

__device__ __forceinline__ __bf16 f2bf(float x) {
    return __builtin_bit_cast(__bf16, __float2bfloat16(x));
}

__device__ __forceinline__ f32x8 wmma_bf16(bf16x16 a, bf16x16 b, f32x8 c) {
    return __builtin_amdgcn_wmma_f32_16x16x32_bf16(false, a, false, b,
                                                   (short)0, c, false, false);
}

// CDNA5 async copy: 16B global -> LDS, tracked by ASYNCcnt (ISA 15.18.3 op 98).
__device__ __forceinline__ void async_copy_b128(void* lds_dst, const void* gsrc) {
    unsigned int l = (unsigned int)(uintptr_t)lds_dst;
    unsigned long long g = (unsigned long long)(uintptr_t)gsrc;
    asm volatile("global_load_async_to_lds_b128 %0, %1, off"
                 :: "v"(l), "v"(g) : "memory");
}
__device__ __forceinline__ void wait_async0() {
    asm volatile("s_wait_asynccnt 0x0" ::: "memory");
}

// 16(x)x32(K) bf16 WMMA operand fragment from a row-major LDS tile.
// Striping: lane = half*16 + x ; element j -> K = (j/8)*16 + half*8 + (j%8).
template<int STRIDE>
__device__ __forceinline__ bf16x16 lds_frag(const __bf16* tile, int rowbase,
                                            int kbase, int lane) {
    const int l16  = lane & 15;
    const int half = (lane >> 4) & 1;
    const __bf16* p = tile + (size_t)(rowbase + l16) * STRIDE + kbase + half * 8;
    bf16x8 lo = *(const bf16x8*)(p);
    bf16x8 hi = *(const bf16x8*)(p + 16);
    return __builtin_shufflevector(lo, hi,
        0,1,2,3,4,5,6,7,8,9,10,11,12,13,14,15);
}

// ---------------------------------------------------------------------------
// Weight pre-convert: fp32 -> bf16, 8 elements per thread.
// ---------------------------------------------------------------------------
__global__ __launch_bounds__(256)
void f32_to_bf16(const float* __restrict__ src, __bf16* __restrict__ dst, int n) {
    int i = (blockIdx.x * 256 + threadIdx.x) * 8;
    if (i < n) {
        float4 a = *(const float4*)(src + i);
        float4 b = *(const float4*)(src + i + 4);
        bf16x8 o;
        o[0]=f2bf(a.x); o[1]=f2bf(a.y); o[2]=f2bf(a.z); o[3]=f2bf(a.w);
        o[4]=f2bf(b.x); o[5]=f2bf(b.y); o[6]=f2bf(b.z); o[7]=f2bf(b.w);
        *(bf16x8*)(dst + i) = o;
    }
}

// ---------------------------------------------------------------------------
// GEMM: C = A[M,K] * W[N,K]^T + bias; W is pre-converted bf16 (async-staged).
//   TIN:  float (convert while staging) or __bf16 (async copy)
//   OMODE 0: fp32 row-major [M,N]
//   OMODE 1: bf16 head-major  out[b][h][s][dk], scaled by oscale
//   OMODE 2: bf16 transposed  out[b][h][dk][s] (vectorized 16B stores)
// Double-buffered LDS, one barrier per K-step.
// ---------------------------------------------------------------------------
#define GT_M 128
#define GT_N 128
#define GT_K 32

template<typename TIN, int OMODE>
__global__ __launch_bounds__(256)
void gemm_bias(const TIN* __restrict__ A,
               const __bf16* __restrict__ W,
               const float* __restrict__ bias,
               void* __restrict__ Cout,
               int M, int N, int K, int sLog, float oscale) {
    __shared__ __attribute__((aligned(16))) __bf16 lA[2][GT_M][GT_K];
    __shared__ __attribute__((aligned(16))) __bf16 lB[2][GT_N][GT_K];

    const int tid  = threadIdx.x;
    const int lane = tid & 31;
    const int wv   = tid >> 5;
    const int l16  = lane & 15;
    const int half = (lane >> 4) & 1;

    const int n0 = blockIdx.x * GT_N;
    const int m0 = blockIdx.y * GT_M;
    const int waveM = wv >> 2;   // 0..1
    const int waveN = wv & 3;    // 0..3

    f32x8 acc[4][2];
#pragma unroll
    for (int i = 0; i < 4; ++i)
#pragma unroll
        for (int j = 0; j < 2; ++j)
            acc[i][j] = f32x8{0.f,0.f,0.f,0.f,0.f,0.f,0.f,0.f};

    // --- staging helper (issues loads/copies into buffer bp for k-slice k0)
    auto stage = [&](int k0, int bp) {
        // Weights: 128x32 bf16 = 512 16B chunks, async copy, 2 per thread.
#pragma unroll
        for (int i = 0; i < 2; ++i) {
            int c   = tid + i * 256;
            int row = c >> 2;
            int c8  = (c & 3) << 3;
            async_copy_b128(&lB[bp][row][c8],
                            W + (size_t)(n0 + row) * K + k0 + c8);
        }
        if constexpr (sizeof(TIN) == 4) {
            // fp32 activations: 1024 float4, 4 per thread, convert to bf16.
#pragma unroll
            for (int i = 0; i < 4; ++i) {
                int f4  = tid + i * 256;
                int row = f4 >> 3;
                int c4  = (f4 & 7) << 2;
                float4 va = *(const float4*)((const float*)A +
                                             (size_t)(m0 + row) * K + k0 + c4);
                __bf16* da = &lA[bp][row][c4];
                da[0]=f2bf(va.x); da[1]=f2bf(va.y);
                da[2]=f2bf(va.z); da[3]=f2bf(va.w);
            }
        } else {
#pragma unroll
            for (int i = 0; i < 2; ++i) {
                int c   = tid + i * 256;
                int row = c >> 2;
                int c8  = (c & 3) << 3;
                async_copy_b128(&lA[bp][row][c8],
                                (const __bf16*)A + (size_t)(m0 + row) * K + k0 + c8);
            }
        }
    };

    stage(0, 0);
    wait_async0();
    __syncthreads();

    int p = 0;
    for (int k0 = 0; k0 < K; k0 += GT_K, p ^= 1) {
        if (k0 + GT_K < K) stage(k0 + GT_K, p ^ 1);

        bf16x16 af[4], bfr[2];
#pragma unroll
        for (int i = 0; i < 4; ++i)
            af[i] = lds_frag<GT_K>(&lA[p][0][0], waveM * 64 + i * 16, 0, lane);
#pragma unroll
        for (int j = 0; j < 2; ++j)
            bfr[j] = lds_frag<GT_K>(&lB[p][0][0], waveN * 32 + j * 16, 0, lane);
#pragma unroll
        for (int i = 0; i < 4; ++i)
#pragma unroll
            for (int j = 0; j < 2; ++j)
                acc[i][j] = wmma_bf16(af[i], bfr[j], acc[i][j]);

        wait_async0();
        __syncthreads();
    }

    // Epilogue. C/D layout: lane = half*16 + n, VGPR r -> row half*8 + r.
    const int sMask = (1 << sLog) - 1;
    const int Hn = N >> 6;
#pragma unroll
    for (int j = 0; j < 2; ++j) {
        const int n = n0 + waveN * 32 + j * 16 + l16;
        const float bv = bias[n];
        const int hh = n >> 6, dk = n & 63;
        if constexpr (OMODE == 2) {
            __bf16* C = (__bf16*)Cout;
#pragma unroll
            for (int i = 0; i < 4; ++i) {
                const int mb = m0 + waveM * 64 + i * 16 + half * 8;
                const int b = mb >> sLog, s = mb & sMask;
                bf16x8 pk;
#pragma unroll
                for (int r = 0; r < 8; ++r)
                    pk[r] = f2bf((acc[i][j][r] + bv) * oscale);
                *(bf16x8*)&C[(((size_t)b * Hn + hh) * 64 + dk) << sLog | s] = pk;
            }
        } else {
#pragma unroll
            for (int i = 0; i < 4; ++i) {
#pragma unroll
                for (int r = 0; r < 8; ++r) {
                    const int m = m0 + waveM * 64 + i * 16 + half * 8 + r;
                    if constexpr (OMODE == 0) {
                        ((float*)Cout)[(size_t)m * N + n] = acc[i][j][r] + bv;
                    } else {
                        const int b = m >> sLog, s = m & sMask;
                        ((__bf16*)Cout)[((((size_t)b * Hn + hh) << sLog) + s) * 64 + dk]
                            = f2bf((acc[i][j][r] + bv) * oscale);
                    }
                }
            }
        }
    }
}

// ---------------------------------------------------------------------------
// Flash attention: grid = (S/128, B*H), 256 threads (8 waves), 64-key blocks,
// double-buffered async LDS staging, WMMA row-sums for the softmax denom.
//   Qh,Kh: bf16 [B][H][S][64] (Q pre-scaled by 1/8);  Vt: bf16 [B][H][64][S]
//   Cx:    bf16 [B][S][D] token-major.
// ---------------------------------------------------------------------------
#define ATT_H  16
#define ATT_DK 64
#define ATT_KB 64

__global__ __launch_bounds__(256)
void flash_attn_bf16(const __bf16* __restrict__ Qh, const __bf16* __restrict__ Kh,
                     const __bf16* __restrict__ Vt, __bf16* __restrict__ Cx,
                     int S, int Dm) {
    __shared__ __attribute__((aligned(16))) __bf16 lK [2][ATT_KB][ATT_DK]; // [key][dk]
    __shared__ __attribute__((aligned(16))) __bf16 lVt[2][ATT_DK][ATT_KB]; // [dk][key]
    __shared__ __attribute__((aligned(16))) __bf16 lP [8][16][ATT_KB];     // per-wave P

    const int tid  = threadIdx.x;
    const int lane = tid & 31;
    const int wv   = tid >> 5;
    const int l16  = lane & 15;
    const int half = (lane >> 4) & 1;

    const int bh = blockIdx.y;
    const int h  = bh & (ATT_H - 1);
    const int b  = bh >> 4;
    const int q0 = blockIdx.x * 128 + wv * 16;

    const __bf16* Qb = Qh + (size_t)bh * S * ATT_DK;   // [S][64]
    const __bf16* Kb = Kh + (size_t)bh * S * ATT_DK;   // [S][64]
    const __bf16* Vb = Vt + (size_t)bh * ATT_DK * S;   // [64][S]

    // Q operand fragments (bf16, pre-scaled by 1/sqrt(dk)).
    bf16x16 aq[2];
    {
        const __bf16* qr = Qb + (size_t)(q0 + l16) * ATT_DK;
#pragma unroll
        for (int t = 0; t < 2; ++t) {
            bf16x8 lo = *(const bf16x8*)(qr + t * 32 + half * 8);
            bf16x8 hi = *(const bf16x8*)(qr + t * 32 + 16 + half * 8);
            aq[t] = __builtin_shufflevector(lo, hi,
                0,1,2,3,4,5,6,7,8,9,10,11,12,13,14,15);
        }
    }

    bf16x16 vones;
#pragma unroll
    for (int i = 0; i < 16; ++i) vones[i] = f2bf(1.0f);

    f32x8 acc[4];
#pragma unroll
    for (int j = 0; j < 4; ++j)
        acc[j] = f32x8{0.f,0.f,0.f,0.f,0.f,0.f,0.f,0.f};
    float mrow[8], lrow[8];
#pragma unroll
    for (int r = 0; r < 8; ++r) { mrow[r] = -INFINITY; lrow[r] = 0.f; }

    // Stage a 64-key block (K: 8KB contiguous, V: 64 rows x 128B): async only.
    auto stage = [&](int kb, int bp) {
#pragma unroll
        for (int i = 0; i < 2; ++i) {
            int c = tid + i * 256;                 // 512 chunks of 16B
            async_copy_b128(&((__bf16*)lK[bp])[c * 8],
                            Kb + (size_t)kb * ATT_DK + c * 8);
        }
#pragma unroll
        for (int i = 0; i < 2; ++i) {
            int c   = tid + i * 256;
            int row = c >> 3;                      // dk
            int c8  = (c & 7) << 3;                // key chunk
            async_copy_b128(&lVt[bp][row][c8], Vb + (size_t)row * S + kb + c8);
        }
    };

    stage(0, 0);
    wait_async0();
    __syncthreads();

    int p = 0;
    for (int kb = 0; kb < S; kb += ATT_KB, p ^= 1) {
        if (kb + ATT_KB < S) stage(kb + ATT_KB, p ^ 1);

        // Scores: four 16q x 16key tiles; K-dim = dk (two chained wmmas each).
        f32x8 s[4];
#pragma unroll
        for (int t = 0; t < 4; ++t) {
            f32x8 z = f32x8{0.f,0.f,0.f,0.f,0.f,0.f,0.f,0.f};
            bf16x16 bk0 = lds_frag<ATT_DK>(&lK[p][0][0], t * 16, 0,  lane);
            bf16x16 bk1 = lds_frag<ATT_DK>(&lK[p][0][0], t * 16, 32, lane);
            z = wmma_bf16(aq[0], bk0, z);
            z = wmma_bf16(aq[1], bk1, z);
            s[t] = z;
        }

        // Online softmax max + rescale (row M = half*8 + r; xor masks 1..8
        // stay inside each 16-lane half). Sums come from WMMA below.
        float alpha[8];
#pragma unroll
        for (int r = 0; r < 8; ++r) {
            float bm = fmaxf(fmaxf(s[0][r], s[1][r]), fmaxf(s[2][r], s[3][r]));
#pragma unroll
            for (int msk = 8; msk >= 1; msk >>= 1)
                bm = fmaxf(bm, __shfl_xor(bm, msk, 32));
            const float mnew = fmaxf(mrow[r], bm);
            alpha[r] = __expf(mrow[r] - mnew);
            mrow[r] = mnew;
#pragma unroll
            for (int t = 0; t < 4; ++t)
                lP[wv][half * 8 + r][t * 16 + l16] = f2bf(__expf(s[t][r] - mnew));
#pragma unroll
            for (int j = 0; j < 4; ++j) acc[j][r] *= alpha[r];
        }

        // P fragments (also used for the row-sum wmma).
        bf16x16 ap0 = lds_frag<ATT_KB>(&lP[wv][0][0], 0, 0,  lane);
        bf16x16 ap1 = lds_frag<ATT_KB>(&lP[wv][0][0], 0, 32, lane);

        // Row sums: P x ones -> every lane's reg r = rowsum(row half*8+r).
        f32x8 rs = f32x8{0.f,0.f,0.f,0.f,0.f,0.f,0.f,0.f};
        rs = wmma_bf16(ap0, vones, rs);
        rs = wmma_bf16(ap1, vones, rs);
#pragma unroll
        for (int r = 0; r < 8; ++r)
            lrow[r] = lrow[r] * alpha[r] + rs[r];

        // O += P(16x64) * V(64x64): 4 dk subtiles x 2 key halves.
#pragma unroll
        for (int j = 0; j < 4; ++j) {
            bf16x16 bv0 = lds_frag<ATT_KB>(&lVt[p][0][0], j * 16, 0,  lane);
            bf16x16 bv1 = lds_frag<ATT_KB>(&lVt[p][0][0], j * 16, 32, lane);
            acc[j] = wmma_bf16(ap0, bv0, acc[j]);
            acc[j] = wmma_bf16(ap1, bv1, acc[j]);
        }

        wait_async0();
        __syncthreads();
    }

    // Epilogue: normalize, write ctx (bf16, token-major).
#pragma unroll
    for (int r = 0; r < 8; ++r) {
        const float inv = 1.0f / lrow[r];
        const int srow = q0 + half * 8 + r;
        __bf16* orow = Cx + ((size_t)b * S + srow) * Dm + h * ATT_DK;
#pragma unroll
        for (int j = 0; j < 4; ++j)
            orow[j * 16 + l16] = f2bf(acc[j][r] * inv);
    }
}

// ---------------------------------------------------------------------------
// Host-side launch.
// ---------------------------------------------------------------------------
extern "C" void kernel_launch(void* const* d_in, const int* in_sizes, int n_in,
                              void* d_out, int out_size, void* d_ws, size_t ws_size,
                              hipStream_t stream) {
    (void)n_in; (void)out_size; (void)ws_size;
    const float* query = (const float*)d_in[0];
    const float* key_  = (const float*)d_in[1];
    const float* value = (const float*)d_in[2];
    const float* Wq = (const float*)d_in[3]; const float* bq = (const float*)d_in[4];
    const float* Wk = (const float*)d_in[5]; const float* bk = (const float*)d_in[6];
    const float* Wv = (const float*)d_in[7]; const float* bv = (const float*)d_in[8];
    const float* Wo = (const float*)d_in[9]; const float* bo = (const float*)d_in[10];
    float* out = (float*)d_out;

    const int D  = 1024;
    const int Bc = 2;
    const int BS = in_sizes[0] / D;  // 4096
    const int S  = BS / Bc;          // 2048
    int sLog = 0; while ((1 << sLog) < S) ++sLog;  // 11

    __bf16* Qh  = (__bf16*)d_ws;                   // [B][H][S][64]
    __bf16* Kh  = Qh  + (size_t)BS * D;
    __bf16* Vt  = Kh  + (size_t)BS * D;            // [B][H][64][S]
    __bf16* Cx  = Vt  + (size_t)BS * D;            // [B][S][D]
    __bf16* Wqb = Cx  + (size_t)BS * D;            // bf16 weights
    __bf16* Wkb = Wqb + (size_t)D * D;
    __bf16* Wvb = Wkb + (size_t)D * D;
    __bf16* Wob = Wvb + (size_t)D * D;

    const int nW = D * D;
    dim3 cg((nW / 8 + 255) / 256);
    f32_to_bf16<<<cg, 256, 0, stream>>>(Wq, Wqb, nW);
    f32_to_bf16<<<cg, 256, 0, stream>>>(Wk, Wkb, nW);
    f32_to_bf16<<<cg, 256, 0, stream>>>(Wv, Wvb, nW);
    f32_to_bf16<<<cg, 256, 0, stream>>>(Wo, Wob, nW);

    dim3 gg(D / GT_N, BS / GT_M);    // (8, 32)
    gemm_bias<float, 1><<<gg, 256, 0, stream>>>(query, Wqb, bq, Qh, BS, D, D, sLog, 0.125f);
    gemm_bias<float, 1><<<gg, 256, 0, stream>>>(key_,  Wkb, bk, Kh, BS, D, D, sLog, 1.0f);
    gemm_bias<float, 2><<<gg, 256, 0, stream>>>(value, Wvb, bv, Vt, BS, D, D, sLog, 1.0f);

    flash_attn_bf16<<<dim3(S / 128, Bc * ATT_H), 256, 0, stream>>>(Qh, Kh, Vt, Cx, S, D);

    gemm_bias<__bf16, 0><<<gg, 256, 0, stream>>>(Cx, Wob, bo, out, BS, D, D, sLog, 1.0f);
}
